// Block_26173530702092
// MI455X (gfx1250) — matrix operand order repaired and checked by reading, not compiled, not run on hardware
//
#include <hip/hip_runtime.h>

// ---------------------------------------------------------------------------
// Transformer block (pre-LN) for MI455X / gfx1250, wave32 + WMMA bf16.
// GEMMs: v_wmma_f32_16x16x32_bf16; double-buffered LDS tiles.
//   A tile:  per-lane global_load_async_to_lds_b128 (ASYNCcnt)
//   W tile:  Tensor Data Mover tensor_load_to_lds (TENSORcnt), wave 0 issues,
//            D# pad feature writes the padded LDS row stride directly.
// Attention: flash-style online softmax, WMMA for Q*K^T and P*V.
// ---------------------------------------------------------------------------

#define BB 8
#define TT 1024
#define CC 1024
#define NHH 16
#define HDD 64
#define FFF 4096
#define MM (BB * TT)          // 8192 rows
#define THREEC (3 * CC)       // 3072

typedef __attribute__((ext_vector_type(16))) __bf16 bf16x16;
typedef __attribute__((ext_vector_type(8)))  float  floatx8;
typedef __attribute__((ext_vector_type(4)))  unsigned int u32x4;
typedef __attribute__((ext_vector_type(8)))  int i32x8;
typedef __attribute__((ext_vector_type(4)))  int i32x4;

union FragU {
    bf16x16 v;
    u32x4   u[2];
    unsigned short s[16];
};
union V8U {
    u32x4 u;
    unsigned short s[8];
};

static __device__ __forceinline__ unsigned short f2bf(float x) {
    unsigned int u = __float_as_uint(x);
    u += 0x7FFFu + ((u >> 16) & 1u);          // round-to-nearest-even
    return (unsigned short)(u >> 16);
}

static __device__ __forceinline__ floatx8 fzero8() {
    floatx8 z;
#pragma unroll
    for (int i = 0; i < 8; ++i) z[i] = 0.0f;
    return z;
}

static __device__ __forceinline__ bf16x16 ld_frag(const unsigned short* p0,
                                                  const unsigned short* p1) {
    FragU f;
    f.u[0] = *(const u32x4*)p0;
    f.u[1] = *(const u32x4*)p1;
    return f.v;
}

static __device__ __forceinline__ floatx8 wmma_bf16(bf16x16 a, bf16x16 b, floatx8 c) {
    return __builtin_amdgcn_wmma_f32_16x16x32_bf16(false, a, false, b,
                                                   (short)0, c, false, false);
}

// Async global -> LDS copy of 16 bytes per lane (ASYNCcnt). LDS byte address
// is the low 32 bits of the generic pointer (flat LDS aperture, ISA 10.2).
static __device__ __forceinline__ void async_ld16(const void* gptr, void* lptr) {
    unsigned lds = (unsigned)(unsigned long long)(size_t)lptr;
    unsigned long long ga = (unsigned long long)(size_t)gptr;
    asm volatile("global_load_async_to_lds_b128 %0, %1, off"
                 :: "v"(lds), "v"(ga) : "memory");
}
template <int N>
static __device__ __forceinline__ void wait_async() {
    asm volatile("s_wait_asynccnt %0" :: "n"(N) : "memory");
}

// TDM: load a [rows x 32] bf16 tile (row stride = K elements in memory) into
// LDS with a 16B pad after every 64B row -> LDS row stride 80B (40 elements).
// D# group0: count=1, lds_addr, 57-bit global_addr, type=2.
// D# group1: data_size=1(2B), pad_enable, pad_interval=3 (16 DW), pad_amount=3
//            (4 DW), tensor_dim0=K, tensor_dim1=rows, tile_dim0=32,
//            tile_dim1=rows, tensor_dim0_stride=K. Groups 2/3: zero (2-D).
// amdgpu-toolchain (clang-23) builtin: 6 args, extra zero-filled i32x8.
static __device__ __forceinline__ void tdm_load_tile(const unsigned short* gbase,
                                                     void* lds, int K, int rows) {
    unsigned long long ga = (unsigned long long)(size_t)gbase;
    unsigned lb = (unsigned)(unsigned long long)(size_t)lds;
    u32x4 g0;
    g0[0] = 1u;                                            // count=1, user desc
    g0[1] = lb;                                            // lds_addr
    g0[2] = (unsigned)(ga & 0xFFFFFFFFull);                // global_addr[31:0]
    g0[3] = (unsigned)((ga >> 32) & 0x01FFFFFFull) | 0x80000000u;  // addr[56:32] | type=2
    i32x8 g1;
    g1[0] = (int)((1u << 16) | (1u << 20) | (3u << 22) | (3u << 25));
    g1[1] = (int)(((unsigned)K & 0xFFFFu) << 16);          // tensor_dim0[15:0]
    g1[2] = (int)((((unsigned)K >> 16) & 0xFFFFu) |
                  (((unsigned)rows & 0xFFFFu) << 16));     // dim0 hi | tensor_dim1 lo
    g1[3] = (int)((32u) << 16);                            // tile_dim0 = 32
    g1[4] = (int)((unsigned)rows & 0xFFFFu);               // tile_dim1
    g1[5] = K;                                             // tensor_dim0_stride
    g1[6] = 0;
    g1[7] = 0;
    i32x4 gz4;
    gz4[0] = 0; gz4[1] = 0; gz4[2] = 0; gz4[3] = 0;
    i32x8 gz8;
#pragma unroll
    for (int i = 0; i < 8; ++i) gz8[i] = 0;
    __builtin_amdgcn_tensor_load_to_lds(g0, g1, gz4, gz4, gz8, 0);
}

// ---------------------------------------------------------------------------
// Weight convert + transpose: f32 W[K][N] -> bf16 Wt[N][K]
// ---------------------------------------------------------------------------
__global__ void wconv_kernel(const float* __restrict__ w,
                             unsigned short* __restrict__ out,
                             int K, int N) {
    long long idx = (long long)blockIdx.x * 256 + threadIdx.x;
    long long total = (long long)K * N;
    if (idx >= total) return;
    int n = (int)(idx / K);
    int k = (int)(idx % K);
    out[(size_t)n * K + k] = f2bf(w[(size_t)k * N + n]);
}

// ---------------------------------------------------------------------------
// LayerNorm: f32 row -> bf16 row
// ---------------------------------------------------------------------------
__global__ void ln_kernel(const float* __restrict__ x,
                          const float* __restrict__ g,
                          const float* __restrict__ b,
                          unsigned short* __restrict__ out) {
    const int row = blockIdx.x;
    const int tid = threadIdx.x;
    const float* xr = x + (size_t)row * CC;
    float s = 0.f, s2 = 0.f;
    for (int i = tid; i < CC; i += 256) {
        float v = xr[i];
        s += v;
        s2 += v * v;
    }
    __shared__ float rs[256];
    __shared__ float rs2[256];
    rs[tid] = s;
    rs2[tid] = s2;
    __syncthreads();
    for (int off = 128; off > 0; off >>= 1) {
        if (tid < off) {
            rs[tid] += rs[tid + off];
            rs2[tid] += rs2[tid + off];
        }
        __syncthreads();
    }
    const float mean = rs[0] * (1.0f / CC);
    const float var  = rs2[0] * (1.0f / CC) - mean * mean;
    const float inv  = rsqrtf(var + 1e-5f);
    for (int i = tid; i < CC; i += 256) {
        float v = (xr[i] - mean) * inv * g[i] + b[i];
        out[(size_t)row * CC + i] = f2bf(v);
    }
}

// ---------------------------------------------------------------------------
// Tiled WMMA GEMM:  out[M,N] = A_bf16[M,K] @ Wt_bf16[N,K]^T + bias (+res)(relu)
// Block tile 128x128x32, 8 waves x (2x4) 16x16 WMMA tiles, double-buffered.
// A tile via per-lane async->LDS; W tile via TDM (wave 0).
// ---------------------------------------------------------------------------
template <bool RELU, bool RES, bool OUTB>
__global__ __launch_bounds__(256) void gemm_kernel(
    const unsigned short* __restrict__ A,
    const unsigned short* __restrict__ Wt,
    const float* __restrict__ bias,
    const float* __restrict__ res,
    float* __restrict__ outF,
    unsigned short* __restrict__ outB,
    int M, int N, int K) {
    constexpr int LS = 40;  // padded LDS row stride (elements) == TDM pad
    __shared__ __align__(16) unsigned short As[2][128 * LS];
    __shared__ __align__(16) unsigned short Bs[2][128 * LS];

    const int tid  = threadIdx.x;
    const int wave = tid >> 5;
    const int lane = tid & 31;
    const int h    = lane >> 4;
    const int l16  = lane & 15;
    const int wm   = wave & 3;   // 4 waves along M
    const int wn   = wave >> 2;  // 2 waves along N
    const int bm   = blockIdx.y * 128;
    const int bn   = blockIdx.x * 128;

    floatx8 acc[2][4];
#pragma unroll
    for (int i = 0; i < 2; ++i)
#pragma unroll
        for (int j = 0; j < 4; ++j) acc[i][j] = fzero8();

    const int srow = tid >> 2;          // staging row 0..63 (x2 chunks)
    const int skc  = (tid & 3) * 8;     // staging k offset

    auto stage = [&](int ks, int p) {
#pragma unroll
        for (int i = 0; i < 2; ++i) {   // A tile: 2 async 16B copies per lane
            int row = srow + i * 64;
            async_ld16(&A[(size_t)(bm + row) * K + ks * 32 + skc],
                       &As[p][row * LS + skc]);
        }
        if (wave == 0)                  // W tile: one TDM DMA per k-step
            tdm_load_tile(&Wt[(size_t)bn * K + ks * 32], &Bs[p][0], K, 128);
    };

    const int nk = K >> 5;
    stage(0, 0);
    for (int kk = 0; kk < nk; ++kk) {
        const int p = kk & 1;
        if (kk + 1 < nk) {
            stage(kk + 1, p ^ 1);
            wait_async<2>();   // prev buffer's A copies done; prefetch in flight
            if (wave == 0) __builtin_amdgcn_s_wait_tensorcnt(1);
        } else {
            wait_async<0>();
            if (wave == 0) __builtin_amdgcn_s_wait_tensorcnt(0);
        }
        __syncthreads();

        bf16x16 af[2];
#pragma unroll
        for (int tm = 0; tm < 2; ++tm) {
            int r = wm * 32 + tm * 16 + l16;
            af[tm] = ld_frag(&As[p][r * LS + h * 8], &As[p][r * LS + 16 + h * 8]);
        }
        bf16x16 bfg[4];
#pragma unroll
        for (int tn = 0; tn < 4; ++tn) {
            int r = wn * 64 + tn * 16 + l16;
            bfg[tn] = ld_frag(&Bs[p][r * LS + h * 8], &Bs[p][r * LS + 16 + h * 8]);
        }
#pragma unroll
        for (int tm = 0; tm < 2; ++tm)
#pragma unroll
            for (int tn = 0; tn < 4; ++tn)
                acc[tm][tn] = wmma_bf16(af[tm], bfg[tn], acc[tm][tn]);
        __syncthreads();   // all waves done reading buf p before it is refilled
    }

    // --- epilogue
#pragma unroll
    for (int tm = 0; tm < 2; ++tm) {
#pragma unroll
        for (int tn = 0; tn < 4; ++tn) {
            int col = bn + wn * 64 + tn * 16 + l16;
            float bcol = bias[col];
#pragma unroll
            for (int r = 0; r < 8; ++r) {
                int row = bm + wm * 32 + tm * 16 + r + 8 * h;
                float v = acc[tm][tn][r] + bcol;
                if (RES) v += res[(size_t)row * N + col];
                if (RELU) v = fmaxf(v, 0.0f);
                if (OUTB)
                    outB[(size_t)row * N + col] = f2bf(v);
                else
                    outF[(size_t)row * N + col] = v;
            }
        }
    }
}

// ---------------------------------------------------------------------------
// Flash attention: block = 128 query rows of one (b, head); 8 waves x 16 rows.
// Double-buffered K/V chunks (K via async global->LDS; V transposed via ds).
// ---------------------------------------------------------------------------
__global__ __launch_bounds__(256) void attn_kernel(
    const unsigned short* __restrict__ qkv,  // bf16 [B,T,3C]
    const int* __restrict__ seq_ls,
    unsigned short* __restrict__ y) {        // bf16 [B,T,C]
    constexpr int KS = 72;  // Ks row stride (elems): [key][hd]
    constexpr int VS = 40;  // Vs row stride: [hd][key]
    constexpr int PS = 40;  // per-wave P scratch stride: [row][key]
    __shared__ __align__(16) unsigned short Ks[2][32 * KS];
    __shared__ __align__(16) unsigned short Vs[2][64 * VS];
    __shared__ __align__(16) unsigned short Ps[8 * 16 * PS];

    const int qt   = blockIdx.x;
    const int head = blockIdx.y;
    const int b    = blockIdx.z;
    const int tid  = threadIdx.x;
    const int wave = tid >> 5;
    const int lane = tid & 31;
    const int h    = lane >> 4;
    const int l16  = lane & 15;
    const int seq_l = seq_ls[b];
    const int qbase = qt * 128 + wave * 16;
    unsigned short* Pw = &Ps[wave * 16 * PS];

    // --- Q fragments (16 rows x 64 hd) stay in registers
    const unsigned short* qrow =
        qkv + ((size_t)(b * TT + qbase + l16)) * THREEC + head * HDD;
    bf16x16 aq[2];
#pragma unroll
    for (int kk = 0; kk < 2; ++kk)
        aq[kk] = ld_frag(qrow + kk * 32 + h * 8, qrow + kk * 32 + 16 + h * 8);

    floatx8 o[4];
#pragma unroll
    for (int t = 0; t < 4; ++t) o[t] = fzero8();
    float mrow[8], lrow[8];
#pragma unroll
    for (int r = 0; r < 8; ++r) {
        mrow[r] = -1e30f;
        lrow[r] = 0.0f;
    }

    const int skey = tid >> 3;          // staging key row 0..31
    const int shd  = (tid & 7) * 8;     // staging hd offset

    auto stage = [&](int kc, int p) {
        // K chunk [32 keys][64 hd]: async global->LDS (1 x 16B per thread)
        async_ld16(qkv + ((size_t)(b * TT + kc + skey)) * THREEC + CC +
                       head * HDD + shd,
                   &Ks[p][skey * KS + shd]);
        // V chunk transposed [64 hd][32 keys]: element transpose via ds stores
        const unsigned short* vp =
            qkv + ((size_t)(b * TT + kc + skey)) * THREEC + 2 * CC + head * HDD + shd;
        V8U vv;
        vv.u = *(const u32x4*)vp;
#pragma unroll
        for (int e = 0; e < 8; ++e) Vs[p][(shd + e) * VS + skey] = vv.s[e];
    };

    const int nchunks = (seq_l + 31) >> 5;
    stage(0, 0);
    for (int ci = 0; ci < nchunks; ++ci) {
        const int p  = ci & 1;
        const int kc = ci * 32;
        if (ci + 1 < nchunks) {
            stage((ci + 1) * 32, p ^ 1);
            wait_async<1>();
        } else {
            wait_async<0>();
        }
        __syncthreads();

        // --- scores S = Q * K^T for two 16-key tiles
        floatx8 s[2];
#pragma unroll
        for (int j = 0; j < 2; ++j) {
            const unsigned short* kr = &Ks[p][(j * 16 + l16) * KS];
            bf16x16 bk0 = ld_frag(kr + h * 8, kr + 16 + h * 8);           // hd 0..31
            bf16x16 bk1 = ld_frag(kr + 32 + h * 8, kr + 32 + 16 + h * 8); // hd 32..63
            floatx8 c = fzero8();
            c = wmma_bf16(aq[0], bk0, c);
            c = wmma_bf16(aq[1], bk1, c);
            s[j] = c;
        }
        // --- scale + key-padding mask (column = key index)
#pragma unroll
        for (int j = 0; j < 2; ++j) {
            bool valid = (kc + j * 16 + l16) < seq_l;
#pragma unroll
            for (int e = 0; e < 8; ++e)
                s[j][e] = valid ? s[j][e] * 0.125f : -1e30f;
        }
        // --- online softmax per row (rows r+8h live in this lane's half)
#pragma unroll
        for (int r = 0; r < 8; ++r) {
            float mv = fmaxf(s[0][r], s[1][r]);
            mv = fmaxf(mv, __shfl_xor(mv, 1, 32));
            mv = fmaxf(mv, __shfl_xor(mv, 2, 32));
            mv = fmaxf(mv, __shfl_xor(mv, 4, 32));
            mv = fmaxf(mv, __shfl_xor(mv, 8, 32));
            float mnew = fmaxf(mrow[r], mv);
            float p0 = __expf(s[0][r] - mnew);
            float p1 = __expf(s[1][r] - mnew);
            s[0][r] = p0;
            s[1][r] = p1;
            float ps = p0 + p1;
            ps += __shfl_xor(ps, 1, 32);
            ps += __shfl_xor(ps, 2, 32);
            ps += __shfl_xor(ps, 4, 32);
            ps += __shfl_xor(ps, 8, 32);
            float corr = __expf(mrow[r] - mnew);
            lrow[r] = lrow[r] * corr + ps;
#pragma unroll
            for (int t = 0; t < 4; ++t) o[t][r] *= corr;
            mrow[r] = mnew;
        }
        // --- transpose P (16x32) into A-fragment layout via per-wave LDS
#pragma unroll
        for (int j = 0; j < 2; ++j)
#pragma unroll
            for (int r = 0; r < 8; ++r)
                Pw[(r + 8 * h) * PS + j * 16 + l16] = f2bf(s[j][r]);
        asm volatile("s_wait_dscnt 0" ::: "memory");  // per-wave LDS RAW fence
        bf16x16 pf = ld_frag(&Pw[l16 * PS + h * 8], &Pw[l16 * PS + 16 + h * 8]);

        // --- O += P(16x32) * V(32x64)
#pragma unroll
        for (int t = 0; t < 4; ++t) {
            const unsigned short* vr = &Vs[p][(t * 16 + l16) * VS];
            bf16x16 bv = ld_frag(vr + h * 8, vr + 16 + h * 8);
            o[t] = wmma_bf16(pf, bv, o[t]);
        }
        __syncthreads();  // all waves done with buf p before it is refilled
    }

    // --- normalize and store y (bf16)
#pragma unroll
    for (int r = 0; r < 8; ++r) {
        float inv = 1.0f / lrow[r];
        int row = qbase + r + 8 * h;
#pragma unroll
        for (int t = 0; t < 4; ++t)
            y[((size_t)(b * TT + row)) * CC + head * HDD + t * 16 + l16] =
                f2bf(o[t][r] * inv);
    }
}

// ---------------------------------------------------------------------------
// Host-side pipeline
// ---------------------------------------------------------------------------
extern "C" void kernel_launch(void* const* d_in, const int* in_sizes, int n_in,
                              void* d_out, int out_size, void* d_ws, size_t ws_size,
                              hipStream_t stream) {
    (void)in_sizes; (void)n_in; (void)out_size; (void)ws_size;
    const float* x      = (const float*)d_in[0];
    const int*   seq_ls = (const int*)d_in[1];
    const float* ln1_g  = (const float*)d_in[2];
    const float* ln1_b  = (const float*)d_in[3];
    const float* w_qkv  = (const float*)d_in[4];
    const float* b_qkv  = (const float*)d_in[5];
    const float* w_proj = (const float*)d_in[6];
    const float* b_proj = (const float*)d_in[7];
    const float* ln2_g  = (const float*)d_in[8];
    const float* ln2_b  = (const float*)d_in[9];
    const float* w_fc   = (const float*)d_in[10];
    const float* b_fc   = (const float*)d_in[11];
    const float* w_fc2  = (const float*)d_in[12];
    const float* b_fc2  = (const float*)d_in[13];
    float* out = (float*)d_out;

    char* ws = (char*)d_ws;
    size_t o_wqkv = 0;                                    // bf16 [3C][C]
    size_t o_wproj = o_wqkv + (size_t)THREEC * CC * 2;    // bf16 [C][C]
    size_t o_wfc   = o_wproj + (size_t)CC * CC * 2;       // bf16 [FF][C]
    size_t o_wfc2  = o_wfc + (size_t)FFF * CC * 2;        // bf16 [C][FF]
    size_t o_act   = o_wfc2 + (size_t)CC * FFF * 2;       // bf16 [M][C]  (h -> y -> h2)
    size_t o_big   = o_act + (size_t)MM * CC * 2;         // bf16 qkv [M][3C] then a [M][FF]
    size_t o_x1    = o_big + (size_t)MM * FFF * 2;        // f32 [M][C]

    unsigned short* wqkvT  = (unsigned short*)(ws + o_wqkv);
    unsigned short* wprojT = (unsigned short*)(ws + o_wproj);
    unsigned short* wfcT   = (unsigned short*)(ws + o_wfc);
    unsigned short* wfc2T  = (unsigned short*)(ws + o_wfc2);
    unsigned short* act_bf = (unsigned short*)(ws + o_act);
    unsigned short* big_bf = (unsigned short*)(ws + o_big);
    float*          x1     = (float*)(ws + o_x1);

    // 1) weights -> bf16 transposed [N][K]
    wconv_kernel<<<((size_t)CC * THREEC + 255) / 256, 256, 0, stream>>>(w_qkv, wqkvT, CC, THREEC);
    wconv_kernel<<<((size_t)CC * CC + 255) / 256, 256, 0, stream>>>(w_proj, wprojT, CC, CC);
    wconv_kernel<<<((size_t)CC * FFF + 255) / 256, 256, 0, stream>>>(w_fc, wfcT, CC, FFF);
    wconv_kernel<<<((size_t)FFF * CC + 255) / 256, 256, 0, stream>>>(w_fc2, wfc2T, FFF, CC);

    // 2) LN1: x -> h (bf16)
    ln_kernel<<<MM, 256, 0, stream>>>(x, ln1_g, ln1_b, act_bf);

    // 3) qkv = h @ w_qkv + b_qkv   (bf16 out)
    gemm_kernel<false, false, true><<<dim3(THREEC / 128, MM / 128), 256, 0, stream>>>(
        act_bf, wqkvT, b_qkv, nullptr, nullptr, big_bf, MM, THREEC, CC);

    // 4) flash attention -> y (bf16, reuses h region)
    attn_kernel<<<dim3(TT / 128, NHH, BB), 256, 0, stream>>>(big_bf, seq_ls, act_bf);

    // 5) x1 = x + y @ w_proj + b_proj  (f32)
    gemm_kernel<false, true, false><<<dim3(CC / 128, MM / 128), 256, 0, stream>>>(
        act_bf, wprojT, b_proj, x, x1, nullptr, MM, CC, CC);

    // 6) LN2: x1 -> h2 (bf16, reuses act region)
    ln_kernel<<<MM, 256, 0, stream>>>(x1, ln2_g, ln2_b, act_bf);

    // 7) a = relu(h2 @ w_fc + b_fc)  (bf16, reuses qkv region)
    gemm_kernel<true, false, true><<<dim3(FFF / 128, MM / 128), 256, 0, stream>>>(
        act_bf, wfcT, b_fc, nullptr, nullptr, big_bf, MM, FFF, CC);

    // 8) out = x1 + a @ w_fc2 + b_fc2  (f32)
    gemm_kernel<false, true, false><<<dim3(CC / 128, MM / 128), 256, 0, stream>>>(
        big_bf, wfc2T, b_fc2, x1, out, nullptr, MM, CC, FFF);
}
